// EVQA_6837587935787
// MI455X (gfx1250) — matrix operand order repaired and checked by reading, not compiled, not run on hardware
//
#include <hip/hip_runtime.h>
#include <hip/hip_bf16.h>

typedef __attribute__((ext_vector_type(16))) _Float16 v16h;
typedef __attribute__((ext_vector_type(8)))  _Float16 v8h;
typedef __attribute__((ext_vector_type(4)))  _Float16 v4h;
typedef __attribute__((ext_vector_type(8)))  float    v8f;

#define B_SZ   4
#define SQ_SZ  2048
#define SKV_SZ 2048
#define D_SZ   1024
#define H_SZ   16
#define DH_SZ  64
#define SCALE_F 0.125f  /* 64^-0.5 */

// ---------------------------------------------------------------------------
// CDNA5 async global->LDS copy (16 bytes per lane), tracked by ASYNCcnt.
// dst: generic pointer into __shared__ (low 32 bits == LDS address)
// ---------------------------------------------------------------------------
__device__ __forceinline__ void async_g2l_b128(void* lds, const void* g) {
    unsigned l = (unsigned)(unsigned long long)lds;
    asm volatile("global_load_async_to_lds_b128 %0, %1, off"
                 :: "v"(l), "v"((unsigned long long)g)
                 : "memory");
}

__device__ __forceinline__ void wait_asynccnt0() {
#if __has_builtin(__builtin_amdgcn_s_wait_asynccnt)
    __builtin_amdgcn_s_wait_asynccnt(0);
#else
    asm volatile("s_wait_asynccnt 0" ::: "memory");
#endif
}

// ---------------------------------------------------------------------------
// fp32 -> f16 conversion (vectorized, B128 global loads)
// ---------------------------------------------------------------------------
__global__ void cvt_f32_to_f16(const float* __restrict__ in,
                               _Float16* __restrict__ out, int n) {
    int i = (blockIdx.x * blockDim.x + threadIdx.x) * 4;
    if (i >= n) return;
    float4 v = *(const float4*)(in + i);
    v4h o;
    o[0] = (_Float16)v.x; o[1] = (_Float16)v.y;
    o[2] = (_Float16)v.z; o[3] = (_Float16)v.w;
    *(v4h*)(out + i) = o;
}

// ---------------------------------------------------------------------------
// Tiled f16 GEMM: C[M,N] = A[M,K] @ B[K,N]
// OUT_MODE 0: f16 output.  OUT_MODE 1: f32 output + bias.
// Block: 256 threads (8 waves). BM=128 BN=128 BK=32.
// Wave tile: 32x64 -> 2(M) x 4(N) frags of v_wmma_f32_16x16x32_f16.
// A tile staged with global_load_async_to_lds_b128; B tile staged with a
// manual N-major transpose (async copies cannot scatter 2-byte elements).
// ---------------------------------------------------------------------------
#define TM 128
#define TN 128
#define TK 32

template <int OUT_MODE>
__global__ __launch_bounds__(256)
void gemm_f16(const _Float16* __restrict__ A, const _Float16* __restrict__ B,
              void* __restrict__ Cout, const float* __restrict__ bias,
              int M, int N, int K) {
    __shared__ _Float16 As[TM][TK + 8];   // row-major (M x K)
    __shared__ _Float16 Bs[TN][TK + 8];   // N-major  (N x K) => contiguous-K frag reads

    const int tid   = threadIdx.x;
    const int lane  = tid & 31;
    const int wave  = tid >> 5;
    const int l16   = lane & 15;
    const int lhalf = lane >> 4;          // 0: lanes 0-15, 1: lanes 16-31
    const int m0    = blockIdx.x * TM;
    const int n0    = blockIdx.y * TN;
    const int wm    = (wave & 3) * 32;    // wave M offset in tile
    const int wn    = (wave >> 2) * 64;   // wave N offset in tile

    v8f zero = {};
    v8f acc[2][4];
#pragma unroll
    for (int i = 0; i < 2; i++)
#pragma unroll
        for (int j = 0; j < 4; j++) acc[i][j] = zero;

    const int k0a = lhalf * 8;            // A frag: K chunks {k0a, k0a+16}

    // per-thread staging coordinates
    const int arow = tid >> 1;
    const int acol = (tid & 1) * 16;
    const int brow = tid >> 3;            // 0..31 (K)
    const int bcol = (tid & 7) * 16;      // 0..112 (N)

    for (int kb = 0; kb < K; kb += TK) {
        const _Float16* gpa = A + (long)(m0 + arow) * K + kb + acol;
        const _Float16* gpb = B + (long)(kb + brow) * N + n0 + bcol;

        // --- stage A tile via async global->LDS (2 x 16B per thread) ---
        async_g2l_b128(&As[arow][acol],     gpa);
        async_g2l_b128(&As[arow][acol + 8], gpa + 8);

        // --- stage B tile transposed: B[kb+r][n0+c] -> Bs[c][r] ---
        {
            v8h x0 = *(const v8h*)gpb;
            v8h x1 = *(const v8h*)(gpb + 8);
#pragma unroll
            for (int i = 0; i < 8; i++) Bs[bcol + i][brow]     = x0[i];
#pragma unroll
            for (int i = 0; i < 8; i++) Bs[bcol + 8 + i][brow] = x1[i];
        }

        // prefetch next K-step tiles into cache while this one computes
        if (kb + TK < K) {
            __builtin_prefetch(gpa + TK, 0, 1);
            __builtin_prefetch(gpb + (long)TK * N, 0, 1);
        }

        wait_asynccnt0();
        __syncthreads();

        // --- fragments ---
        v16h af[2], bf[4];
#pragma unroll
        for (int i = 0; i < 2; i++) {
            const _Float16* ap = &As[wm + i * 16 + l16][0];
            v8h lo = *(const v8h*)(ap + k0a);
            v8h hi = *(const v8h*)(ap + k0a + 16);
#pragma unroll
            for (int t = 0; t < 8; t++) { af[i][t] = lo[t]; af[i][8 + t] = hi[t]; }
        }
#pragma unroll
        for (int j = 0; j < 4; j++) {
            const _Float16* bp = &Bs[wn + j * 16 + l16][lhalf * 8];
            v8h b0 = *(const v8h*)bp;
            v8h b1 = *(const v8h*)(bp + 16);
#pragma unroll
            for (int t = 0; t < 8; t++) { bf[j][t] = b0[t]; bf[j][8 + t] = b1[t]; }
        }
#pragma unroll
        for (int i = 0; i < 2; i++)
#pragma unroll
            for (int j = 0; j < 4; j++)
                acc[i][j] = __builtin_amdgcn_wmma_f32_16x16x32_f16(
                    false, af[i], false, bf[j], (short)0, acc[i][j], false, false);
        __syncthreads();
    }

    // --- epilogue ---
#pragma unroll
    for (int i = 0; i < 2; i++)
#pragma unroll
        for (int j = 0; j < 4; j++)
#pragma unroll
            for (int r = 0; r < 8; r++) {
                int row = m0 + wm + i * 16 + r + 8 * lhalf;
                int col = n0 + wn + j * 16 + l16;
                if (OUT_MODE == 0) {
                    ((_Float16*)Cout)[(long)row * N + col] = (_Float16)acc[i][j][r];
                } else {
                    ((float*)Cout)[(long)row * N + col] = acc[i][j][r] + bias[col];
                }
            }
}

// ---------------------------------------------------------------------------
// Flash attention: one block per (b, h, 128 q-rows). 8 waves x 16 q-rows.
// Q frags persist in registers; K/V chunks of 64 keys staged in LDS
// (K via async global->LDS, V via manual transpose).
// S = Q.K^T via WMMA -> online softmax -> P through per-wave LDS -> PV WMMA.
// ---------------------------------------------------------------------------
__global__ __launch_bounds__(256)
void attn_fused(const _Float16* __restrict__ Q, const _Float16* __restrict__ Kc,
                const _Float16* __restrict__ V, _Float16* __restrict__ O) {
    __shared__ _Float16 Ks[64][72];          // [key][d]   (natural: B frags for QK^T)
    __shared__ _Float16 Vs[64][72];          // [d][key]   (transposed: B frags for PV)
    __shared__ _Float16 Pw[8][16][72];       // per-wave P staging (C-layout -> A-layout)

    const int tid   = threadIdx.x;
    const int lane  = tid & 31;
    const int wave  = tid >> 5;
    const int l16   = lane & 15;
    const int lhalf = lane >> 4;

    const int nqb = SQ_SZ / 128;
    const int qb  = blockIdx.x % nqb;
    const int h   = (blockIdx.x / nqb) % H_SZ;
    const int b   = blockIdx.x / (nqb * H_SZ);

    const int qrow = qb * 128 + wave * 16 + l16;

    // preload Q A-frags (16 rows x 64 dh -> 2 frags of 16x32)
    v16h qf[2];
    {
        const _Float16* qp = Q + (long)(b * SQ_SZ + qrow) * D_SZ + h * DH_SZ;
#pragma unroll
        for (int ks = 0; ks < 2; ks++) {
            int k0 = ks * 32 + lhalf * 8;
            v8h lo = *(const v8h*)(qp + k0);
            v8h hi = *(const v8h*)(qp + k0 + 16);
#pragma unroll
            for (int t = 0; t < 8; t++) { qf[ks][t] = lo[t]; qf[ks][8 + t] = hi[t]; }
        }
    }

    v8f zero = {};
    v8f o_acc[4];
#pragma unroll
    for (int j = 0; j < 4; j++) o_acc[j] = zero;
    float mrow[8], lrow[8];
#pragma unroll
    for (int r = 0; r < 8; r++) { mrow[r] = -3.0e38f; lrow[r] = 0.0f; }

    // per-thread staging coordinates (64 rows x 64 halfs, 4 threads per row)
    const int srow = tid >> 2;
    const int sd0  = (tid & 3) * 16;
    const _Float16* gpK = Kc + (long)(b * SKV_SZ + srow) * D_SZ + h * DH_SZ + sd0;
    const _Float16* gpV = V  + (long)(b * SKV_SZ + srow) * D_SZ + h * DH_SZ + sd0;

    for (int kc = 0; kc < SKV_SZ; kc += 64) {
        __syncthreads();
        const _Float16* gk = gpK + (long)kc * D_SZ;
        const _Float16* gv = gpV + (long)kc * D_SZ;

        // --- stage K chunk [key][d] via async global->LDS ---
        async_g2l_b128(&Ks[srow][sd0],     gk);
        async_g2l_b128(&Ks[srow][sd0 + 8], gk + 8);

        // --- stage V chunk transposed [d][key] (manual 2-byte scatter) ---
        {
            v8h x0 = *(const v8h*)gv;
            v8h x1 = *(const v8h*)(gv + 8);
#pragma unroll
            for (int i = 0; i < 8; i++) Vs[sd0 + i][srow]     = x0[i];
#pragma unroll
            for (int i = 0; i < 8; i++) Vs[sd0 + 8 + i][srow] = x1[i];
        }

        // prefetch next chunk into cache while this chunk computes
        if (kc + 64 < SKV_SZ) {
            __builtin_prefetch(gk + (long)64 * D_SZ, 0, 1);
            __builtin_prefetch(gv + (long)64 * D_SZ, 0, 1);
        }

        wait_asynccnt0();
        __syncthreads();

        // --- S = scale * Q.K^T : 4 N-frags of 16x16 ---
        v8f s[4];
#pragma unroll
        for (int j = 0; j < 4; j++) {
            s[j] = zero;
#pragma unroll
            for (int ks = 0; ks < 2; ks++) {
                const _Float16* bp = &Ks[j * 16 + l16][ks * 32 + lhalf * 8];
                v8h b0 = *(const v8h*)bp;
                v8h b1 = *(const v8h*)(bp + 16);
                v16h bf;
#pragma unroll
                for (int t = 0; t < 8; t++) { bf[t] = b0[t]; bf[8 + t] = b1[t]; }
                s[j] = __builtin_amdgcn_wmma_f32_16x16x32_f16(
                    false, qf[ks], false, bf, (short)0, s[j], false, false);
            }
            s[j] = s[j] * SCALE_F;
        }

        // --- online softmax (rows live across 16-lane groups) ---
#pragma unroll
        for (int r = 0; r < 8; r++) {
            float mx = fmaxf(fmaxf(s[0][r], s[1][r]), fmaxf(s[2][r], s[3][r]));
#pragma unroll
            for (int off = 1; off < 16; off <<= 1)
                mx = fmaxf(mx, __shfl_xor(mx, off, 16));
            float mnew = fmaxf(mrow[r], mx);
            float corr = __expf(mrow[r] - mnew);
            mrow[r] = mnew;
            float ps = 0.0f;
#pragma unroll
            for (int j = 0; j < 4; j++) {
                float p = __expf(s[j][r] - mnew);
                s[j][r] = p;
                ps += p;
            }
#pragma unroll
            for (int off = 1; off < 16; off <<= 1)
                ps += __shfl_xor(ps, off, 16);
            lrow[r] = lrow[r] * corr + ps;
#pragma unroll
            for (int j = 0; j < 4; j++) o_acc[j][r] *= corr;
        }

        // --- stage P (C-layout) into per-wave LDS ---
#pragma unroll
        for (int j = 0; j < 4; j++)
#pragma unroll
            for (int r = 0; r < 8; r++)
                Pw[wave][r + 8 * lhalf][j * 16 + l16] = (_Float16)s[j][r];
        __syncthreads();

        // --- read P as A-frags, accumulate O += P.V ---
        v16h pf[2];
#pragma unroll
        for (int ks = 0; ks < 2; ks++) {
            const _Float16* pp = &Pw[wave][l16][ks * 32 + lhalf * 8];
            v8h lo = *(const v8h*)pp;
            v8h hi = *(const v8h*)(pp + 16);
#pragma unroll
            for (int t = 0; t < 8; t++) { pf[ks][t] = lo[t]; pf[ks][8 + t] = hi[t]; }
        }
#pragma unroll
        for (int j = 0; j < 4; j++) {
#pragma unroll
            for (int ks = 0; ks < 2; ks++) {
                const _Float16* bp = &Vs[j * 16 + l16][ks * 32 + lhalf * 8];
                v8h b0 = *(const v8h*)bp;
                v8h b1 = *(const v8h*)(bp + 16);
                v16h bf;
#pragma unroll
                for (int t = 0; t < 8; t++) { bf[t] = b0[t]; bf[8 + t] = b1[t]; }
                o_acc[j] = __builtin_amdgcn_wmma_f32_16x16x32_f16(
                    false, pf[ks], false, bf, (short)0, o_acc[j], false, false);
            }
        }
    }

    // --- normalize and store merged-head f16 output [B*SQ, D] ---
#pragma unroll
    for (int r = 0; r < 8; r++) {
        float inv = 1.0f / lrow[r];
        int row = qb * 128 + wave * 16 + r + 8 * lhalf;
#pragma unroll
        for (int j = 0; j < 4; j++) {
            int col = h * DH_SZ + j * 16 + l16;
            O[(long)(b * SQ_SZ + row) * D_SZ + col] = (_Float16)(o_acc[j][r] * inv);
        }
    }
}

// ---------------------------------------------------------------------------
// Host-side launch
// ---------------------------------------------------------------------------
extern "C" void kernel_launch(void* const* d_in, const int* in_sizes, int n_in,
                              void* d_out, int out_size, void* d_ws, size_t ws_size,
                              hipStream_t stream) {
    const float* hs = (const float*)d_in[0];
    const float* es = (const float*)d_in[1];
    const float* Wq = (const float*)d_in[2];
    const float* Wk = (const float*)d_in[3];
    const float* Wv = (const float*)d_in[4];
    const float* Wo = (const float*)d_in[5];
    const float* bo = (const float*)d_in[6];
    float* out = (float*)d_out;

    const size_t ACT = (size_t)B_SZ * SQ_SZ * D_SZ;   // 8,388,608 elems
    const size_t WEL = (size_t)D_SZ * D_SZ;           // 1,048,576 elems

    char* ws = (char*)d_ws;
    size_t off = 0;
    _Float16* h16  = (_Float16*)(ws + off); off += ACT * 2;  // reused as attn output
    _Float16* e16  = (_Float16*)(ws + off); off += ACT * 2;
    _Float16* q16  = (_Float16*)(ws + off); off += ACT * 2;
    _Float16* k16  = (_Float16*)(ws + off); off += ACT * 2;
    _Float16* v16  = (_Float16*)(ws + off); off += ACT * 2;
    _Float16* wq16 = (_Float16*)(ws + off); off += WEL * 2;
    _Float16* wk16 = (_Float16*)(ws + off); off += WEL * 2;
    _Float16* wv16 = (_Float16*)(ws + off); off += WEL * 2;
    _Float16* wo16 = (_Float16*)(ws + off); off += WEL * 2;

    // 1) fp32 -> f16 conversions
    cvt_f32_to_f16<<<(int)(ACT / 4 / 256), 256, 0, stream>>>(hs, h16, (int)ACT);
    cvt_f32_to_f16<<<(int)(ACT / 4 / 256), 256, 0, stream>>>(es, e16, (int)ACT);
    cvt_f32_to_f16<<<(int)(WEL / 4 / 256), 256, 0, stream>>>(Wq, wq16, (int)WEL);
    cvt_f32_to_f16<<<(int)(WEL / 4 / 256), 256, 0, stream>>>(Wk, wk16, (int)WEL);
    cvt_f32_to_f16<<<(int)(WEL / 4 / 256), 256, 0, stream>>>(Wv, wv16, (int)WEL);
    cvt_f32_to_f16<<<(int)(WEL / 4 / 256), 256, 0, stream>>>(Wo, wo16, (int)WEL);

    // 2) projections: Q = hs@Wq, K = es@Wk, V = es@Wv
    dim3 gg(B_SZ * SQ_SZ / TM, D_SZ / TN);
    gemm_f16<0><<<gg, 256, 0, stream>>>(h16, wq16, q16, nullptr,
                                        B_SZ * SQ_SZ, D_SZ, D_SZ);
    gemm_f16<0><<<gg, 256, 0, stream>>>(e16, wk16, k16, nullptr,
                                        B_SZ * SQ_SZ, D_SZ, D_SZ);
    gemm_f16<0><<<gg, 256, 0, stream>>>(e16, wv16, v16, nullptr,
                                        B_SZ * SQ_SZ, D_SZ, D_SZ);

    // 3) fused attention (softmax(QK^T*scale))V -> h16 (reused, merged heads)
    attn_fused<<<B_SZ * H_SZ * (SQ_SZ / 128), 256, 0, stream>>>(q16, k16, v16, h16);

    // 4) output projection with bias (f32 output)
    gemm_f16<1><<<gg, 256, 0, stream>>>(h16, wo16, out, bo,
                                        B_SZ * SQ_SZ, D_SZ, D_SZ);
}
